// MemoryBank_17102559772928
// MI455X (gfx1250) — matrix operand order repaired and checked by reading, not compiled, not run on hardware
//
#include <hip/hip_runtime.h>

// ---------------------------------------------------------------------------
// Fixed problem dimensions (from the reference)
// ---------------------------------------------------------------------------
#define NTOK  2048      // 2*1024 tokens
#define EMB   512
#define HID   512
#define G4H   2048      // 4*HID gate width
#define NQ    256       // queries
#define TOPK  8

typedef __bf16 bf16;
typedef __bf16 v8bf  __attribute__((ext_vector_type(8)));
typedef __bf16 v16bf __attribute__((ext_vector_type(16)));
typedef float  v8f   __attribute__((ext_vector_type(8)));

__device__ __forceinline__ bf16 to_bf16(float f) {
    union { float f; unsigned u; } v; v.f = f;
    unsigned r = v.u + 0x7FFFu + ((v.u >> 16) & 1u);   // round-to-nearest-even
    unsigned short h = (unsigned short)(r >> 16);
    return __builtin_bit_cast(bf16, h);
}
__device__ __forceinline__ float sigmoidf_(float x) {
    return 1.0f / (1.0f + __expf(-x));
}

// ---------------------------------------------------------------------------
// 1. Stable compaction: selected token indices (score > 0.5), original order.
//    Single workgroup, 1024 threads, 2 elements each, LDS Hillis-Steele scan.
// ---------------------------------------------------------------------------
__global__ __launch_bounds__(1024)
void scan_select(const float* __restrict__ scores,
                 int* __restrict__ sel_idx, int* __restrict__ n_sel) {
    __shared__ int s[1024];
    int tid = threadIdx.x;
    int m0 = scores[2 * tid]     > 0.5f ? 1 : 0;
    int m1 = scores[2 * tid + 1] > 0.5f ? 1 : 0;
    s[tid] = m0 + m1;
    __syncthreads();
    for (int off = 1; off < 1024; off <<= 1) {
        int v   = s[tid];
        int add = (tid >= off) ? s[tid - off] : 0;
        __syncthreads();
        s[tid] = v + add;
        __syncthreads();
    }
    int incl = s[tid];
    int excl = incl - (m0 + m1);
    if (m0) sel_idx[excl]      = 2 * tid;
    if (m1) sel_idx[excl + m0] = 2 * tid + 1;
    if (tid == 1023) *n_sel = incl;
}

// ---------------------------------------------------------------------------
// 2. Gather selected keys (-> bf16 A-operand) and values (f32 bank values).
//    Rows >= n_sel are zeroed so the downstream GEMM stays defined.
// ---------------------------------------------------------------------------
__global__ __launch_bounds__(256)
void gather_rows(const float* __restrict__ keys, const float* __restrict__ values,
                 const int* __restrict__ sel_idx, const int* __restrict__ n_sel,
                 bf16* __restrict__ ks, float* __restrict__ vs) {
    int t  = blockIdx.x;
    int ns = *n_sel;
    if (t < ns) {
        int src = sel_idx[t];
        for (int i = threadIdx.x; i < EMB; i += 256) {
            ks[(size_t)t * EMB + i] = to_bf16(keys[(size_t)src * EMB + i]);
            vs[(size_t)t * EMB + i] = values[(size_t)src * EMB + i];
        }
    } else {
        for (int i = threadIdx.x; i < EMB; i += 256)
            ks[(size_t)t * EMB + i] = to_bf16(0.0f);
    }
}

// ---------------------------------------------------------------------------
// 3. f32 -> bf16 conversion (weights) and bias sum b_ih + b_hh
// ---------------------------------------------------------------------------
__global__ __launch_bounds__(256)
void cvt_bf16(const float* __restrict__ in, bf16* __restrict__ out, int n) {
    for (int i = blockIdx.x * blockDim.x + threadIdx.x; i < n; i += gridDim.x * blockDim.x)
        out[i] = to_bf16(in[i]);
}
__global__ __launch_bounds__(256)
void add_vec(const float* __restrict__ a, const float* __restrict__ b,
             float* __restrict__ out, int n) {
    for (int i = blockIdx.x * blockDim.x + threadIdx.x; i < n; i += gridDim.x * blockDim.x)
        out[i] = a[i] + b[i];
}

// ---------------------------------------------------------------------------
// 4. WMMA GEMM, 32x32 macro-tile per wave (2x2 register blocking):
//      C[M,N] = A[M,K] * B[N,K]^T (+ bias[N])
//    bf16 operands, f32 accumulate. Per K-step: 2 A-frags + 2 B-frags feed
//    4 v_wmma_f32_16x16x32_bf16 -> 32 B loaded per WMMA (2x reuse vs naive).
//    Operand layouts per CDNA5 ISA (wave32):
//      A lane: row = mt*16 + lane%16; K-segs [k0+(lane/16)*8, +8) and
//              [k0+16+(lane/16)*8, +8)
//      B lane: row (=N col) = nt*16 + lane%16; K-segs [k0+(lane/16)*8, +8) and
//              [k0+16+(lane/16)*8, +8)
//      C lane: col = nt*16 + lane%16; rows r + 8*(lane/16), r = 0..7
//    If nvalid != nullptr, columns >= *nvalid are stored as -inf (sims mask).
//    Requires M % 32 == 0, N % 32 == 0, K % 32 == 0.
// ---------------------------------------------------------------------------
__global__ __launch_bounds__(256)
void wmma_gemm(const bf16* __restrict__ A, const bf16* __restrict__ B,
               const float* __restrict__ bias, float* __restrict__ C,
               int M, int N, int K, const int* __restrict__ nvalid) {
    int lane = threadIdx.x & 31;
    int wave = (blockIdx.x * blockDim.x + threadIdx.x) >> 5;
    int Mt = M >> 5, Nt = N >> 5;          // 32x32 macro-tiles
    if (wave >= Mt * Nt) return;
    int tm = wave / Nt, tn = wave % Nt;
    int r16 = lane & 15;
    int hi  = lane >> 4;

    const bf16* Arow0 = A + (size_t)(tm * 32 + r16) * K;        // M-subtile 0
    const bf16* Arow1 = Arow0 + (size_t)16 * K;                 // M-subtile 1
    const bf16* Brow0 = B + (size_t)(tn * 32 + r16) * K;        // N-subtile 0
    const bf16* Brow1 = Brow0 + (size_t)16 * K;                 // N-subtile 1

    float bi0 = bias ? bias[tn * 32 + r16]      : 0.0f;
    float bi1 = bias ? bias[tn * 32 + 16 + r16] : 0.0f;
    v8f acc00, acc01, acc10, acc11;
#pragma unroll
    for (int r = 0; r < 8; ++r) {
        acc00[r] = bi0; acc01[r] = bi1;
        acc10[r] = bi0; acc11[r] = bi1;
    }

    int aoff0 = hi * 8;        // first K-segment offset within step
    int boff  = hi * 8;
    for (int k0 = 0; k0 < K; k0 += 32) {
        v8bf a0l = *(const v8bf*)(Arow0 + k0 + aoff0);
        v8bf a0h = *(const v8bf*)(Arow0 + k0 + 16 + aoff0);
        v8bf a1l = *(const v8bf*)(Arow1 + k0 + aoff0);
        v8bf a1h = *(const v8bf*)(Arow1 + k0 + 16 + aoff0);
        v8bf b0l = *(const v8bf*)(Brow0 + k0 + boff);
        v8bf b0h = *(const v8bf*)(Brow0 + k0 + 16 + boff);
        v8bf b1l = *(const v8bf*)(Brow1 + k0 + boff);
        v8bf b1h = *(const v8bf*)(Brow1 + k0 + 16 + boff);
        v16bf a0 = __builtin_shufflevector(a0l, a0h, 0,1,2,3,4,5,6,7,8,9,10,11,12,13,14,15);
        v16bf a1 = __builtin_shufflevector(a1l, a1h, 0,1,2,3,4,5,6,7,8,9,10,11,12,13,14,15);
        v16bf b0 = __builtin_shufflevector(b0l, b0h, 0,1,2,3,4,5,6,7,8,9,10,11,12,13,14,15);
        v16bf b1 = __builtin_shufflevector(b1l, b1h, 0,1,2,3,4,5,6,7,8,9,10,11,12,13,14,15);
        acc00 = __builtin_amdgcn_wmma_f32_16x16x32_bf16(false, a0, false, b0, (short)0, acc00, false, false);
        acc01 = __builtin_amdgcn_wmma_f32_16x16x32_bf16(false, a0, false, b1, (short)0, acc01, false, false);
        acc10 = __builtin_amdgcn_wmma_f32_16x16x32_bf16(false, a1, false, b0, (short)0, acc10, false, false);
        acc11 = __builtin_amdgcn_wmma_f32_16x16x32_bf16(false, a1, false, b1, (short)0, acc11, false, false);
    }

    int col0 = tn * 32 + r16;
    int col1 = col0 + 16;
    int nv   = nvalid ? *nvalid : 0x7fffffff;
    bool inv0 = col0 >= nv;
    bool inv1 = col1 >= nv;
#pragma unroll
    for (int r = 0; r < 8; ++r) {
        int m0 = tm * 32 + r + 8 * hi;
        int m1 = m0 + 16;
        C[(size_t)m0 * N + col0] = inv0 ? -__builtin_inff() : acc00[r];
        C[(size_t)m0 * N + col1] = inv1 ? -__builtin_inff() : acc01[r];
        C[(size_t)m1 * N + col0] = inv0 ? -__builtin_inff() : acc10[r];
        C[(size_t)m1 * N + col1] = inv1 ? -__builtin_inff() : acc11[r];
    }
}

// ---------------------------------------------------------------------------
// 5. Sequential LSTM over the compacted sequence (single persistent WG).
//    Input gates XG[t, 0..2047] precomputed (includes b_ih+b_hh).
//    Thread tid owns gate rows tid and tid+1024; h lives in LDS, c in regs.
//    W_hh is bf16 (2 MB) -> stays resident in the 192 MB L2. The next
//    timestep's XG row is prefetched (global_prefetch_b8) to overlap the
//    gate fetch with the current matvec.
// ---------------------------------------------------------------------------
__global__ __launch_bounds__(1024)
void lstm_seq(const float* __restrict__ XG, const bf16* __restrict__ Whh,
              const int* __restrict__ n_sel_p, bf16* __restrict__ hs) {
    __shared__ float h_lds[HID];
    __shared__ float g_lds[G4H];
    int tid = threadIdx.x;
    int ns  = *n_sel_p;
    if (tid < HID) h_lds[tid] = 0.0f;
    float c = 0.0f;
    __syncthreads();

    const bf16* w0 = Whh + (size_t)tid * HID;
    const bf16* w1 = Whh + (size_t)(tid + 1024) * HID;

    for (int t = 0; t < ns; ++t) {
        float a0 = XG[(size_t)t * G4H + tid];
        float a1 = XG[(size_t)t * G4H + tid + 1024];
        if (t + 1 < ns) {
            __builtin_prefetch(&XG[(size_t)(t + 1) * G4H + tid], 0, 0);
            __builtin_prefetch(&XG[(size_t)(t + 1) * G4H + tid + 1024], 0, 0);
        }
        for (int e = 0; e < HID; e += 8) {
            v8bf wv0 = *(const v8bf*)(w0 + e);
            v8bf wv1 = *(const v8bf*)(w1 + e);
#pragma unroll
            for (int q = 0; q < 8; ++q) {
                float hv = h_lds[e + q];
                a0 += (float)wv0[q] * hv;
                a1 += (float)wv1[q] * hv;
            }
        }
        g_lds[tid]        = a0;
        g_lds[tid + 1024] = a1;
        __syncthreads();                      // gates done; h reads finished
        if (tid < HID) {
            float ig = sigmoidf_(g_lds[tid]);
            float fg = sigmoidf_(g_lds[HID + tid]);
            float gg = tanhf(g_lds[2 * HID + tid]);
            float og = sigmoidf_(g_lds[3 * HID + tid]);
            c = fg * c + ig * gg;
            float h = og * tanhf(c);
            h_lds[tid] = h;
            hs[(size_t)t * HID + tid] = to_bf16(h);
        }
        __syncthreads();                      // new h visible for next step
    }
}

// ---------------------------------------------------------------------------
// 6. Row L2-normalize (f32 in) -> bf16 out, one 256-thread block per row.
// ---------------------------------------------------------------------------
__global__ __launch_bounds__(256)
void normalize_rows(const float* __restrict__ in, bf16* __restrict__ out) {
    __shared__ float red[256];
    int row = blockIdx.x, tid = threadIdx.x;
    const float* r = in + (size_t)row * EMB;
    float ss = 0.0f;
    for (int i = tid; i < EMB; i += 256) { float v = r[i]; ss += v * v; }
    red[tid] = ss;
    __syncthreads();
    for (int off = 128; off > 0; off >>= 1) {
        if (tid < off) red[tid] += red[tid + off];
        __syncthreads();
    }
    float inv = 1.0f / fmaxf(sqrtf(red[0]), 1e-12f);
    for (int i = tid; i < EMB; i += 256)
        out[(size_t)row * EMB + i] = to_bf16(r[i] * inv);
}

// ---------------------------------------------------------------------------
// 7. Top-8 per query: sims row cached in LDS, 8 argmax rounds w/ masking.
//    Tie-break = smallest index (matches lax.top_k first-occurrence).
// ---------------------------------------------------------------------------
__global__ __launch_bounds__(256)
void topk8(const float* __restrict__ sims, int* __restrict__ tki) {
    __shared__ float sv[NTOK];
    __shared__ float rv[256];
    __shared__ int   ri[256];
    int q = blockIdx.x, tid = threadIdx.x;
    for (int i = tid; i < NTOK; i += 256) sv[i] = sims[(size_t)q * NTOK + i];
    __syncthreads();
    for (int k = 0; k < TOPK; ++k) {
        float bv = -__builtin_inff(); int bi = NTOK;
        for (int i = tid; i < NTOK; i += 256) {
            float v = sv[i];
            if (v > bv) { bv = v; bi = i; }
        }
        rv[tid] = bv; ri[tid] = bi;
        __syncthreads();
        for (int off = 128; off > 0; off >>= 1) {
            if (tid < off) {
                bool take = (rv[tid + off] > rv[tid]) ||
                            (rv[tid + off] == rv[tid] && ri[tid + off] < ri[tid]);
                if (take) { rv[tid] = rv[tid + off]; ri[tid] = ri[tid + off]; }
            }
            __syncthreads();
        }
        if (tid == 0) {
            int w = ri[0] < NTOK ? ri[0] : 0;
            tki[q * TOPK + k] = w;
            sv[w] = -__builtin_inff();
        }
        __syncthreads();
    }
}

// ---------------------------------------------------------------------------
// 8. Gather output rows: out[q,k,:] = bank_values[tki[q,k], :]
// ---------------------------------------------------------------------------
__global__ __launch_bounds__(256)
void gather_out(const float* __restrict__ vs, const int* __restrict__ tki,
                float* __restrict__ out) {
    int b   = blockIdx.x;           // q*TOPK + k  (2048 blocks)
    int idx = tki[b];
    const float* src = vs + (size_t)idx * EMB;
    float* dst = out + (size_t)b * EMB;
    for (int i = threadIdx.x; i < EMB; i += 256) dst[i] = src[i];
}

// ---------------------------------------------------------------------------
// Launch
// ---------------------------------------------------------------------------
extern "C" void kernel_launch(void* const* d_in, const int* in_sizes, int n_in,
                              void* d_out, int out_size, void* d_ws, size_t ws_size,
                              hipStream_t stream) {
    const float* keys    = (const float*)d_in[0];
    const float* values  = (const float*)d_in[1];
    const float* scores  = (const float*)d_in[2];
    const float* queries = (const float*)d_in[3];
    const float* W_ih    = (const float*)d_in[6];
    const float* W_hh    = (const float*)d_in[7];
    const float* b_ih    = (const float*)d_in[8];
    const float* b_hh    = (const float*)d_in[9];
    const float* W_out   = (const float*)d_in[10];
    const float* b_out   = (const float*)d_in[11];
    float* out = (float*)d_out;

    // workspace carve-up (256B aligned)
    size_t off = 0;
    auto alloc = [&](size_t bytes) -> char* {
        off = (off + 255) & ~(size_t)255;
        char* p = (char*)d_ws + off;
        off += bytes;
        return p;
    };
    int*   sel_idx = (int*)  alloc(NTOK * sizeof(int));
    int*   n_sel   = (int*)  alloc(sizeof(int));
    bf16*  ks      = (bf16*) alloc((size_t)NTOK * EMB * sizeof(bf16));
    float* vs      = (float*)alloc((size_t)NTOK * EMB * sizeof(float));
    bf16*  Wih_b   = (bf16*) alloc((size_t)G4H * EMB * sizeof(bf16));
    bf16*  Whh_b   = (bf16*) alloc((size_t)G4H * HID * sizeof(bf16));
    bf16*  Wout_b  = (bf16*) alloc((size_t)EMB * HID * sizeof(bf16));
    float* bsum    = (float*)alloc(G4H * sizeof(float));
    float* XG      = (float*)alloc((size_t)NTOK * G4H * sizeof(float));
    bf16*  hs      = (bf16*) alloc((size_t)NTOK * HID * sizeof(bf16));
    float* memout  = (float*)alloc((size_t)NTOK * EMB * sizeof(float));
    bf16*  kn      = (bf16*) alloc((size_t)NTOK * EMB * sizeof(bf16));
    bf16*  qn      = (bf16*) alloc((size_t)NQ * EMB * sizeof(bf16));
    float* sims    = (float*)alloc((size_t)NQ * NTOK * sizeof(float));
    int*   tki     = (int*)  alloc(NQ * TOPK * sizeof(int));
    (void)ws_size; (void)out_size; (void)n_in; (void)in_sizes;

    // 1. selection scan
    scan_select<<<1, 1024, 0, stream>>>(scores, sel_idx, n_sel);

    // 2. gather + convert selected keys/values
    gather_rows<<<NTOK, 256, 0, stream>>>(keys, values, sel_idx, n_sel, ks, vs);

    // 3. weight conversion + bias sum
    cvt_bf16<<<1024, 256, 0, stream>>>(W_ih, Wih_b, G4H * EMB);
    cvt_bf16<<<1024, 256, 0, stream>>>(W_hh, Whh_b, G4H * HID);
    cvt_bf16<<<256, 256, 0, stream>>>(W_out, Wout_b, EMB * HID);
    add_vec<<<8, 256, 0, stream>>>(b_ih, b_hh, bsum, G4H);

    // 4. input-gate GEMM: XG[2048,2048] = ks * W_ih^T + (b_ih + b_hh)
    {
        int tiles = (NTOK / 32) * (G4H / 32);          // 64 * 64
        wmma_gemm<<<(tiles + 7) / 8, 256, 0, stream>>>(
            ks, Wih_b, bsum, XG, NTOK, G4H, EMB, nullptr);
    }

    // 5. recurrent LSTM
    lstm_seq<<<1, 1024, 0, stream>>>(XG, Whh_b, n_sel, hs);

    // 6. output projection: memout[2048,512] = hs * W_out^T + b_out
    {
        int tiles = (NTOK / 32) * (EMB / 32);          // 64 * 16
        wmma_gemm<<<(tiles + 7) / 8, 256, 0, stream>>>(
            hs, Wout_b, b_out, memout, NTOK, EMB, HID, nullptr);
    }

    // 7. normalize bank keys and queries
    normalize_rows<<<NTOK, 256, 0, stream>>>(memout, kn);
    normalize_rows<<<NQ, 256, 0, stream>>>(queries, qn);

    // 8. cosine sims: sims[256,2048] = qn * kn^T, cols >= n_sel -> -inf
    {
        int tiles = (NQ / 32) * (NTOK / 32);           // 8 * 64
        wmma_gemm<<<(tiles + 7) / 8, 256, 0, stream>>>(
            qn, kn, nullptr, sims, NQ, NTOK, EMB, n_sel);
    }

    // 9. top-8 and gather output
    topk8<<<NQ, 256, 0, stream>>>(sims, tki);
    gather_out<<<NQ * TOPK, 256, 0, stream>>>(vs, tki, out);
}
